// DeepGraphConvolution_30150670418536
// MI455X (gfx1250) — matrix operand order repaired and checked by reading, not compiled, not run on hardware
//
#include <hip/hip_runtime.h>
#include <hip/hip_bf16.h>

#define DIM 256
#define TILE_K 32

typedef __attribute__((ext_vector_type(16))) __bf16 v16bf;
typedef __attribute__((ext_vector_type(8)))  __bf16 v8bf;
typedef __attribute__((ext_vector_type(8)))  float  v8f;

// ---- fp32 -> bf16 split helpers (round-to-nearest-even) ----
__device__ __forceinline__ unsigned short bf16_rn_bits(float f) {
    unsigned int u = __builtin_bit_cast(unsigned int, f);
    return (unsigned short)((u + 0x7FFFu + ((u >> 16) & 1u)) >> 16);
}
__device__ __forceinline__ float bf16_bits_to_f32(unsigned short h) {
    unsigned int u = ((unsigned int)h) << 16;
    return __builtin_bit_cast(float, u);
}

// ---------------------------------------------------------------------------
// Kernel 0: one-pass fp32 -> (bf16 hi, bf16 lo) split.  hi = rne(f),
// lo = rne(f - hi).  Each thread converts 4 floats -> 8B hi + 8B lo.
// ---------------------------------------------------------------------------
__global__ __launch_bounds__(256) void split_bf16_kernel(const float* __restrict__ src,
                                                         unsigned short* __restrict__ hi,
                                                         unsigned short* __restrict__ lo,
                                                         int n4) {
    int i = blockIdx.x * blockDim.x + threadIdx.x;
    if (i >= n4) return;
    float4 f = ((const float4*)src)[i];
    ushort4 h, l;
    h.x = bf16_rn_bits(f.x); l.x = bf16_rn_bits(f.x - bf16_bits_to_f32(h.x));
    h.y = bf16_rn_bits(f.y); l.y = bf16_rn_bits(f.y - bf16_bits_to_f32(h.y));
    h.z = bf16_rn_bits(f.z); l.z = bf16_rn_bits(f.z - bf16_bits_to_f32(h.z));
    h.w = bf16_rn_bits(f.w); l.w = bf16_rn_bits(f.w - bf16_bits_to_f32(h.w));
    ((ushort4*)hi)[i] = h;
    ((ushort4*)lo)[i] = l;
}

// ---------------------------------------------------------------------------
// Kernel 1: y2 = x2 (copy) and zero the y1 accumulator region of d_out.
// ---------------------------------------------------------------------------
__global__ __launch_bounds__(256) void init_kernel(const float* __restrict__ x2,
                                                   float* __restrict__ y2,
                                                   float* __restrict__ y1acc,
                                                   int n4) {
    int i = blockIdx.x * blockDim.x + threadIdx.x;
    if (i >= n4) return;
    ((float4*)y2)[i] = ((const float4*)x2)[i];
    ((float4*)y1acc)[i] = make_float4(0.f, 0.f, 0.f, 0.f);
}

// ---------------------------------------------------------------------------
// Kernel 2: hidden = x2 @ W^T + b via bf16x3 split-precision WMMA.
// One wave per 16x16 output tile; operands are pre-split packed bf16, so
// every WMMA source register block is a single contiguous b128 load.
// ---------------------------------------------------------------------------
__device__ __forceinline__ v16bf cat8(v8bf a, v8bf b) {
    return __builtin_shufflevector(a, b, 0, 1, 2, 3, 4, 5, 6, 7,
                                         8, 9, 10, 11, 12, 13, 14, 15);
}

__global__ __launch_bounds__(256) void gemm_bf16x3_kernel(
        const __bf16* __restrict__ x2h, const __bf16* __restrict__ x2l,
        const __bf16* __restrict__ Wh,  const __bf16* __restrict__ Wl,
        const float* __restrict__ bias, float* __restrict__ hidden,
        int n_nodes) {
    const int lane  = threadIdx.x & 31;
    const int wave  = threadIdx.x >> 5;
    const int wglob = blockIdx.x * 8 + wave;
    const int row_tile = wglob >> 4;     // 16 column tiles per row tile (DIM/16)
    const int col_tile = wglob & 15;
    const int row0 = row_tile * 16;
    if (row0 >= n_nodes) return;
    const int col0 = col_tile * 16;

    const int m    = lane & 15;          // A: row M ; B: col N
    const int half = lane >> 4;
    const int ka   = half * 8;           // A K-base: VGPR0-3 chunk; +16 for VGPR4-7
    const int kb   = half * 8;           // B K-base: VGPR0-3 chunk; +16 for VGPR4-7

    int ar = row0 + m;
    if (ar >= n_nodes) ar = n_nodes - 1;           // clamp (tail tile safety)
    const __bf16* arh = x2h + (size_t)ar * DIM;
    const __bf16* arl = x2l + (size_t)ar * DIM;
    const __bf16* brh = Wh + (size_t)(col0 + m) * DIM;
    const __bf16* brl = Wl + (size_t)(col0 + m) * DIM;

    v8f acc = {};

    for (int kk = 0; kk < DIM; kk += TILE_K) {
        // A: two contiguous 8-element chunks per lane (K = ka..ka+7, ka+16..ka+23)
        v16bf ahi = cat8(*(const v8bf*)(arh + kk + ka),
                         *(const v8bf*)(arh + kk + ka + 16));
        v16bf alo = cat8(*(const v8bf*)(arl + kk + ka),
                         *(const v8bf*)(arl + kk + ka + 16));
        // B: two contiguous 8-element chunks per lane (K = kb..kb+7, kb+16..kb+23)
        v16bf bhi = cat8(*(const v8bf*)(brh + kk + kb),
                         *(const v8bf*)(brh + kk + kb + 16));
        v16bf blo = cat8(*(const v8bf*)(brl + kk + kb),
                         *(const v8bf*)(brl + kk + kb + 16));

        // bf16x3: hi*hi + hi*lo + lo*hi (lo*lo ~ 2^-16, dropped)
        acc = __builtin_amdgcn_wmma_f32_16x16x32_bf16(false, ahi, false, bhi,
                                                      (short)0, acc, false, false);
        acc = __builtin_amdgcn_wmma_f32_16x16x32_bf16(false, ahi, false, blo,
                                                      (short)0, acc, false, false);
        acc = __builtin_amdgcn_wmma_f32_16x16x32_bf16(false, alo, false, bhi,
                                                      (short)0, acc, false, false);
    }

    // C/D layout: VGPR r -> row r (lanes 0-15) or r+8 (lanes 16-31), col = lane%16
    const int   ocol = col0 + m;
    const float bv   = bias[ocol];
#pragma unroll
    for (int r = 0; r < 8; ++r) {
        const int orow = row0 + r + half * 8;
        if (orow < n_nodes)
            hidden[(size_t)orow * DIM + ocol] = acc[r] + bv;
    }
}

// ---------------------------------------------------------------------------
// Kernel 3: scatter-add  support[rows[e]] += vals[e] * hidden[cols[e]]
// One wave per edge; each lane covers two float4s of the 256-float row.
// hidden (51 MB) stays resident in the 192 MB L2.
// ---------------------------------------------------------------------------
__global__ __launch_bounds__(256) void edge_scatter_kernel(const int* __restrict__ rows,
                                                           const int* __restrict__ cols,
                                                           const float* __restrict__ vals,
                                                           const float* __restrict__ hidden,
                                                           float* __restrict__ support,
                                                           int n_edges) {
    const int t    = blockIdx.x * blockDim.x + threadIdx.x;
    const int e    = t >> 5;
    const int lane = t & 31;
    if (e >= n_edges) return;

    const int   r = rows[e];
    const int   c = cols[e];
    const float v = vals[e];

    const float4* hsrc = (const float4*)(hidden + (size_t)c * DIM);
    float*        dst  = support + (size_t)r * DIM;

#pragma unroll
    for (int j = 0; j < 2; ++j) {
        const int i4 = lane + j * 32;     // 64 float4 per row
        float4 h = hsrc[i4];
        const int d0 = i4 * 4;
        __hip_atomic_fetch_add(dst + d0 + 0, v * h.x, __ATOMIC_RELAXED, __HIP_MEMORY_SCOPE_AGENT);
        __hip_atomic_fetch_add(dst + d0 + 1, v * h.y, __ATOMIC_RELAXED, __HIP_MEMORY_SCOPE_AGENT);
        __hip_atomic_fetch_add(dst + d0 + 2, v * h.z, __ATOMIC_RELAXED, __HIP_MEMORY_SCOPE_AGENT);
        __hip_atomic_fetch_add(dst + d0 + 3, v * h.w, __ATOMIC_RELAXED, __HIP_MEMORY_SCOPE_AGENT);
    }
}

// ---------------------------------------------------------------------------
// Kernel 4: y1 = relu(acc) + x1  (in place on the accumulator region)
// ---------------------------------------------------------------------------
__global__ __launch_bounds__(256) void finalize_kernel(const float* __restrict__ x1,
                                                       float* __restrict__ y1,
                                                       int n4) {
    int i = blockIdx.x * blockDim.x + threadIdx.x;
    if (i >= n4) return;
    float4 a = ((const float4*)y1)[i];
    float4 x = ((const float4*)x1)[i];
    float4 o;
    o.x = fmaxf(a.x, 0.f) + x.x;
    o.y = fmaxf(a.y, 0.f) + x.y;
    o.z = fmaxf(a.z, 0.f) + x.z;
    o.w = fmaxf(a.w, 0.f) + x.w;
    ((float4*)y1)[i] = o;
}

extern "C" void kernel_launch(void* const* d_in, const int* in_sizes, int n_in,
                              void* d_out, int out_size, void* d_ws, size_t ws_size,
                              hipStream_t stream) {
    const float* x1    = (const float*)d_in[0];
    const float* x2    = (const float*)d_in[1];
    const int*   rows  = (const int*)d_in[2];
    const int*   cols  = (const int*)d_in[3];
    const float* evals = (const float*)d_in[4];
    const float* W     = (const float*)d_in[5];
    const float* bias  = (const float*)d_in[6];

    const int n_nodes = in_sizes[0] / DIM;
    const int n_edges = in_sizes[2];
    const size_t nd   = (size_t)n_nodes * DIM;

    float* y2 = (float*)d_out;                    // output 0: y2 = x2
    float* y1 = y2 + nd;                          // output 1: y1

    // workspace layout: hidden | x2_hi | x2_lo | W_hi | W_lo
    char* ws = (char*)d_ws;
    float*          hidden = (float*)ws;                  ws += nd * sizeof(float);
    unsigned short* x2h    = (unsigned short*)ws;         ws += nd * sizeof(unsigned short);
    unsigned short* x2l    = (unsigned short*)ws;         ws += nd * sizeof(unsigned short);
    unsigned short* Wh     = (unsigned short*)ws;         ws += (size_t)DIM * DIM * sizeof(unsigned short);
    unsigned short* Wl     = (unsigned short*)ws;

    // 0) split x2 and W into bf16 hi/lo (streaming, one pass)
    {
        const int n4 = (int)(nd / 4);
        split_bf16_kernel<<<(n4 + 255) / 256, 256, 0, stream>>>(x2, x2h, x2l, n4);
        const int w4 = (DIM * DIM) / 4;
        split_bf16_kernel<<<(w4 + 255) / 256, 256, 0, stream>>>(W, Wh, Wl, w4);
    }
    // 1) y2 = x2 ; y1 accumulator = 0
    {
        const int n4 = (int)(nd / 4);
        init_kernel<<<(n4 + 255) / 256, 256, 0, stream>>>(x2, y2, y1, n4);
    }
    // 2) hidden = x2 @ W^T + b (WMMA bf16x3, pre-split operands)
    {
        const int row_tiles = (n_nodes + 15) / 16;
        const int waves     = row_tiles * (DIM / 16);   // one wave per 16x16 tile
        const int blocks    = (waves + 7) / 8;          // 8 waves (256 thr) per block
        gemm_bf16x3_kernel<<<blocks, 256, 0, stream>>>((const __bf16*)x2h, (const __bf16*)x2l,
                                                       (const __bf16*)Wh,  (const __bf16*)Wl,
                                                       bias, hidden, n_nodes);
    }
    // 3) edge gather/scale/scatter-add into y1 accumulator
    {
        const long long threads = (long long)n_edges * 32;
        const int blocks = (int)((threads + 255) / 256);
        edge_scatter_kernel<<<blocks, 256, 0, stream>>>(rows, cols, evals, hidden, y1, n_edges);
    }
    // 4) y1 = relu(acc) + x1
    {
        const int n4 = (int)(nd / 4);
        finalize_kernel<<<(n4 + 255) / 256, 256, 0, stream>>>(x1, y1, n4);
    }
}
